// Transformer_LoRA_54245436949095
// MI455X (gfx1250) — matrix operand order repaired and checked by reading, not compiled, not run on hardware
//
#include <hip/hip_runtime.h>

typedef __attribute__((ext_vector_type(16))) _Float16 v16h;
typedef __attribute__((ext_vector_type(8)))  _Float16 v8h;
typedef __attribute__((ext_vector_type(4)))  _Float16 v4h;
typedef __attribute__((ext_vector_type(8)))  float    v8f;
typedef __attribute__((ext_vector_type(4)))  unsigned int u32x4;
typedef __attribute__((ext_vector_type(8)))  int      i32x8;
typedef __attribute__((ext_vector_type(4)))  int      i32x4;

#if defined(__has_builtin)
#  if __has_builtin(__builtin_amdgcn_tensor_load_to_lds) && __has_builtin(__builtin_amdgcn_s_wait_tensorcnt)
#    define HAVE_TDM 1
#  endif
#endif
#ifndef HAVE_TDM
#  define HAVE_TDM 0
#endif

static __device__ __forceinline__ v16h cat8(v8h lo, v8h hi) {
  return __builtin_shufflevector(lo, hi, 0,1,2,3,4,5,6,7,8,9,10,11,12,13,14,15);
}

// ---------------------------------------------------------------------------
// fp32 -> f16 conversion (weights)
// ---------------------------------------------------------------------------
__global__ void k_cvt_f16(const float* __restrict__ src, _Float16* __restrict__ dst, int n) {
  int i = blockIdx.x * blockDim.x + threadIdx.x;
  if (i < n) dst[i] = (_Float16)src[i];
}

// ---------------------------------------------------------------------------
// fused (optional add-pos) + LayerNorm -> f16, plus after_A = h @ lora_A^T
// one 256-thread block per row of C=1024
// ---------------------------------------------------------------------------
static __device__ __forceinline__ float block_sum_256(float v, float* red) {
  __syncthreads();
  int lane = threadIdx.x & 31, w = threadIdx.x >> 5;
  #pragma unroll
  for (int o = 16; o >= 1; o >>= 1) v += __shfl_xor(v, o, 32);
  if (lane == 0) red[w] = v;
  __syncthreads();
  return red[0]+red[1]+red[2]+red[3]+red[4]+red[5]+red[6]+red[7];
}

__global__ void k_add_ln(const float* __restrict__ xin, const float* __restrict__ pos,
                         const float* __restrict__ g, const float* __restrict__ bta,
                         float* __restrict__ xsave, _Float16* __restrict__ hout,
                         const float* __restrict__ loraA, float* __restrict__ afterA,
                         int mode) {
  __shared__ float red[8];
  __shared__ float lsum[8][8];
  const int C = 1024;
  int row = blockIdx.x;
  int t = threadIdx.x;
  const float* xr = xin + (size_t)row * C;
  float4 v = *(const float4*)(xr + t * 4);
  if (mode == 1) {
    float4 p = *(const float4*)(pos + (size_t)row * C + t * 4);
    v.x += p.x; v.y += p.y; v.z += p.z; v.w += p.w;
    *(float4*)(xsave + (size_t)row * C + t * 4) = v;
  }
  float mean = block_sum_256(v.x + v.y + v.z + v.w, red) * (1.0f / C);
  float d0 = v.x - mean, d1 = v.y - mean, d2 = v.z - mean, d3 = v.w - mean;
  float var = block_sum_256(d0*d0 + d1*d1 + d2*d2 + d3*d3, red) * (1.0f / C);
  float inv = rsqrtf(var + 1e-5f);
  float4 gg = *(const float4*)(g + t * 4);
  float4 bb = *(const float4*)(bta + t * 4);
  float h0 = d0 * inv * gg.x + bb.x;
  float h1 = d1 * inv * gg.y + bb.y;
  float h2 = d2 * inv * gg.z + bb.z;
  float h3 = d3 * inv * gg.w + bb.w;
  _Float16* hr = hout + (size_t)row * C + t * 4;
  hr[0] = (_Float16)h0; hr[1] = (_Float16)h1; hr[2] = (_Float16)h2; hr[3] = (_Float16)h3;

  if (mode == 1) {
    int lane = threadIdx.x & 31, w = threadIdx.x >> 5;
    #pragma unroll
    for (int j = 0; j < 8; ++j) {
      const float* la = loraA + (size_t)j * C + t * 4;
      float s = h0 * la[0] + h1 * la[1] + h2 * la[2] + h3 * la[3];
      #pragma unroll
      for (int o = 16; o >= 1; o >>= 1) s += __shfl_xor(s, o, 32);
      if (lane == 0) lsum[w][j] = s;
    }
    __syncthreads();
    if (threadIdx.x < 8) {
      float s = 0.f;
      #pragma unroll
      for (int w2 = 0; w2 < 8; ++w2) s += lsum[w2][threadIdx.x];
      afterA[(size_t)row * 8 + threadIdx.x] = s;
    }
  }
}

// ---------------------------------------------------------------------------
// TDM: load a 32-row x 32-col f16 tile (row stride K elements) into LDS.
// D# built per CDNA5 ISA ch.8: group0 {count=1, lds_addr, global_addr, type=2},
// group1 {data_size=2B, tensor_dim0=K, tensor_dim1=M, tile 32x32, stride0=K}.
// Groups 2/3 unused (2-D tile); this toolchain's builtin takes 6 args.
// ---------------------------------------------------------------------------
#if HAVE_TDM
static __device__ __forceinline__ void tdm_load_tile32x32(const _Float16* gptr,
                                                          unsigned lds_addr,
                                                          int K, int M) {
  unsigned long long ga = (unsigned long long)(size_t)gptr;
  u32x4 g0;
  g0[0] = 1u;                                        // count=1, user mode
  g0[1] = lds_addr;                                  // LDS byte address
  g0[2] = (unsigned)(ga & 0xffffffffu);              // global_addr[31:0]
  g0[3] = (unsigned)((ga >> 32) & 0x1ffffffu) | (2u << 30); // addr[56:32] | type=2
  i32x8 g1;
  g1[0] = 0x00010000;                                // wg_mask=0, data_size=1 (2B)
  g1[1] = (int)(((unsigned)K & 0xffffu) << 16);      // tensor_dim0[15:0] @ [31:16]
  g1[2] = (int)((((unsigned)K >> 16) & 0xffffu) | (((unsigned)M & 0xffffu) << 16));
  g1[3] = (int)((((unsigned)M >> 16) & 0xffffu) | (32u << 16)); // tile_dim0=32
  g1[4] = 32;                                        // tile_dim1=32, tile_dim2=0
  g1[5] = K;                                         // tensor_dim0_stride[31:0]
  g1[6] = 0;
  g1[7] = 0;
  i32x4 z4 = {};
  i32x8 z8 = {};
  __builtin_amdgcn_tensor_load_to_lds(g0, g1, z4, z4, z8, 0);
}
#endif

// ---------------------------------------------------------------------------
// WMMA GEMM: Out[M,Nt] = A_f16[M,K] @ W_f16[Nt,K]^T + bias, + epilogue
//   mode 1: qkv (+ LoRA rank-4 on q/v thirds) -> f16
//   mode 2: proj + residual -> f32
//   mode 3: GELU (exact) -> f16
//   mode 4: mlp2 + residual -> f32 (final output)
// block = 256 threads (8 waves); block tile = 32 rows x 512 cols.
// A k-slab (32x32) staged in LDS via TDM double-buffer (or cooperative copy),
// shared by all 8 waves; each wave: 2 row-tiles x 4 col-tiles = 8 wmma/k-step.
// ---------------------------------------------------------------------------
__global__ void k_gemm_wmma(const _Float16* __restrict__ A,
                            const _Float16* __restrict__ W,
                            const float* __restrict__ bias,
                            int M, int Nt, int K, int mode,
                            const float* __restrict__ afterA,
                            const float* __restrict__ loraB,
                            const float* __restrict__ resid,
                            float* __restrict__ outF,
                            _Float16* __restrict__ outH) {
  __shared__ _Float16 ldsA[2][32 * 32];
  int wave = threadIdx.x >> 5;
  int lane = threadIdx.x & 31;
  int half = lane >> 4;
  int l16  = lane & 15;
  int rbase = blockIdx.x * 32;
  int nbase = blockIdx.y * 512 + wave * 64;

  v8f acc[2][4] = {};
  const _Float16* Aslab = A + (size_t)rbase * K;

#if HAVE_TDM
  unsigned ldsbase = (unsigned)(size_t)&ldsA[0][0];
  if (wave == 0) tdm_load_tile32x32(Aslab, ldsbase, K, M);
#endif

  for (int k = 0; k < K; k += 32) {
#if HAVE_TDM
    int p = (k >> 5) & 1;
    __syncthreads();                       // WAR: prev compute done before overwrite
    if (wave == 0) {
      if (k + 32 < K) {
        tdm_load_tile32x32(Aslab + (k + 32), ldsbase + (unsigned)((p ^ 1) * 32 * 32 * 2), K, M);
        __builtin_amdgcn_s_wait_tensorcnt(1);   // current tile landed
      } else {
        __builtin_amdgcn_s_wait_tensorcnt(0);
      }
    }
    __syncthreads();                       // tile visible to all waves
#else
    int p = 0;
    __syncthreads();
    {
      int r  = threadIdx.x >> 3;           // 0..31
      int c4 = (threadIdx.x & 7) * 4;      // 0..28
      *(v4h*)&ldsA[0][r * 32 + c4] = *(const v4h*)(Aslab + (size_t)r * K + k + c4);
    }
    __syncthreads();
#endif
    const _Float16* ap = &ldsA[p][0];
    v16h a0 = *(const v16h*)(ap + l16 * 32 + 16 * half);
    v16h a1 = *(const v16h*)(ap + (16 + l16) * 32 + 16 * half);
    #pragma unroll
    for (int t = 0; t < 4; ++t) {
      const _Float16* Wrow = W + (size_t)(nbase + t * 16 + l16) * K + k + 16 * half;
      v16h bf = *(const v16h*)Wrow;
      __builtin_prefetch(Wrow + 128, 0, 1);
      acc[0][t] = __builtin_amdgcn_wmma_f32_16x16x32_f16(
          false, a0, false, bf, (short)0, acc[0][t], false, false);
      acc[1][t] = __builtin_amdgcn_wmma_f32_16x16x32_f16(
          false, a1, false, bf, (short)0, acc[1][t], false, false);
    }
  }

  #pragma unroll
  for (int rt = 0; rt < 2; ++rt) {
    #pragma unroll
    for (int t = 0; t < 4; ++t) {
      int col = nbase + t * 16 + l16;
      float bv = bias[col];
      #pragma unroll
      for (int i = 0; i < 8; ++i) {
        int row = rbase + rt * 16 + i + 8 * half;
        float v = acc[rt][t][i] + bv;
        size_t oidx = (size_t)row * Nt + col;
        if (mode == 1) {
          const int Cc = 1024;
          if (col < Cc) {                  // q third
            const float* aA = afterA + (size_t)row * 8;
            const float* lb = loraB + (size_t)col * 4;
            v += 0.25f * (aA[0]*lb[0] + aA[1]*lb[1] + aA[2]*lb[2] + aA[3]*lb[3]);
          } else if (col >= 2 * Cc) {      // v third
            const float* aA = afterA + (size_t)row * 8 + 4;
            const float* lb = loraB + (size_t)(Cc + col - 2 * Cc) * 4;
            v += 0.25f * (aA[0]*lb[0] + aA[1]*lb[1] + aA[2]*lb[2] + aA[3]*lb[3]);
          }
          outH[oidx] = (_Float16)v;
        } else if (mode == 2 || mode == 4) {
          outF[oidx] = resid[oidx] + v;
        } else {                           // mode 3: exact GELU
          float gv = 0.5f * v * (1.0f + erff(v * 0.70710678118f));
          outH[oidx] = (_Float16)gv;
        }
      }
    }
  }
}

// ---------------------------------------------------------------------------
// flash attention: 4 waves per block share one (b,h); each wave owns a 16-row
// query tile. Per 32-key tile: V staged (transposed) once into shared LDS by
// all 128 threads; each wave: 4 wmma for S=QK^T, online softmax via half-wave
// shuffles, P->LDS transpose to A-layout, 4 wmma for O += P V.
// ---------------------------------------------------------------------------
__global__ void k_flash_attn(const _Float16* __restrict__ qkv, _Float16* __restrict__ o) {
  __shared__ _Float16 ldsVt[64 * 32];     // [dh][key-in-tile], shared by 4 waves
  __shared__ _Float16 ldsP[4][16 * 32];   // per-wave P tile
  const int C = 1024, LD = 3072, N = 2048;
  int tid = threadIdx.x;
  int wave = tid >> 5, lane = tid & 31;
  int half = lane >> 4, l16 = lane & 15;
  int qg = blockIdx.x, hh = blockIdx.y, bb = blockIdx.z;
  const _Float16* base = qkv + (size_t)bb * N * LD;
  int qrow = (qg * 4 + wave) * 16;

  const _Float16* qp = base + (size_t)(qrow + l16) * LD + hh * 64;
  v16h qa0 = *(const v16h*)(qp + 16 * half);
  v16h qa1 = *(const v16h*)(qp + 32 + 16 * half);

  v8f oacc[4] = {};
  float mrun[8], lrun[8];
  #pragma unroll
  for (int i = 0; i < 8; ++i) { mrun[i] = -1e30f; lrun[i] = 0.f; }
  const float scale = 0.125f; // Dh^-0.5

  for (int kt = 0; kt < N; kt += 32) {
    __syncthreads();                      // prev iteration's ldsVt reads done
    // cooperative transposed V stage: thread -> key = kt + tid/4, dh chunk = (tid&3)*16
    {
      int kk = tid >> 2;
      int d0 = (tid & 3) * 16;
      const _Float16* vp = base + (size_t)(kt + kk) * LD + 2 * C + hh * 64 + d0;
      v8h va = *(const v8h*)vp;
      v8h vb = *(const v8h*)(vp + 8);
      #pragma unroll
      for (int e = 0; e < 8; ++e) {
        ldsVt[(d0 + e) * 32 + kk]     = va[e];
        ldsVt[(d0 + 8 + e) * 32 + kk] = vb[e];
      }
    }
    // S = Q K^T (two 16-key sub-tiles; K rows contiguous -> direct B frags)
    v8f s[2] = {};
    #pragma unroll
    for (int t = 0; t < 2; ++t) {
      const _Float16* kp = base + (size_t)(kt + t * 16 + l16) * LD + C + hh * 64;
      v16h kb0 = *(const v16h*)(kp + 16 * half);
      v16h kb1 = *(const v16h*)(kp + 32 + 16 * half);
      s[t] = __builtin_amdgcn_wmma_f32_16x16x32_f16(false, qa0, false, kb0, (short)0, s[t], false, false);
      s[t] = __builtin_amdgcn_wmma_f32_16x16x32_f16(false, qa1, false, kb1, (short)0, s[t], false, false);
    }
    // online softmax; a 16-wide row of the C tile lives in one 16-lane half
    #pragma unroll
    for (int i = 0; i < 8; ++i) {
      float v0 = s[0][i] * scale, v1 = s[1][i] * scale;
      float mx = fmaxf(v0, v1);
      #pragma unroll
      for (int o2 = 8; o2 >= 1; o2 >>= 1) mx = fmaxf(mx, __shfl_xor(mx, o2, 32));
      float mnew = fmaxf(mrun[i], mx);
      float corr = __expf(mrun[i] - mnew);
      float p0 = __expf(v0 - mnew), p1 = __expf(v1 - mnew);
      float ps = p0 + p1;
      #pragma unroll
      for (int o2 = 8; o2 >= 1; o2 >>= 1) ps += __shfl_xor(ps, o2, 32);
      lrun[i] = lrun[i] * corr + ps;
      mrun[i] = mnew;
      #pragma unroll
      for (int t = 0; t < 4; ++t) oacc[t][i] *= corr;
      int prow = i + 8 * half;
      ldsP[wave][prow * 32 + l16]      = (_Float16)p0;
      ldsP[wave][prow * 32 + 16 + l16] = (_Float16)p1;
    }
    __syncthreads();                      // V tile fully staged
    // P as A fragment (own-wave DS ops are in order)
    const _Float16* pp = &ldsP[wave][(size_t)l16 * 32];
    v16h pf = *(const v16h*)(pp + 16 * half);
    // O += P @ V (V columns contiguous in ldsVt)
    #pragma unroll
    for (int t = 0; t < 4; ++t) {
      const _Float16* vtp = ldsVt + (size_t)(t * 16 + l16) * 32 + 16 * half;
      v16h vb = *(const v16h*)vtp;
      oacc[t] = __builtin_amdgcn_wmma_f32_16x16x32_f16(false, pf, false, vb, (short)0, oacc[t], false, false);
    }
  }
  // normalize rows, store O as f16 (input to proj GEMM)
  #pragma unroll
  for (int t = 0; t < 4; ++t) {
    #pragma unroll
    for (int i = 0; i < 8; ++i) {
      int row = qrow + i + 8 * half;
      float ov = oacc[t][i] / lrun[i];
      o[((size_t)(bb * N + row)) * C + hh * 64 + t * 16 + l16] = (_Float16)ov;
    }
  }
}

// ---------------------------------------------------------------------------
extern "C" void kernel_launch(void* const* d_in, const int* in_sizes, int n_in,
                              void* d_out, int out_size, void* d_ws, size_t ws_size,
                              hipStream_t stream) {
  const float* x      = (const float*)d_in[0];
  const float* pos    = (const float*)d_in[1];
  const float* Wqkv   = (const float*)d_in[2];
  const float* b_qkv  = (const float*)d_in[3];
  const float* loraA  = (const float*)d_in[4];
  const float* loraB  = (const float*)d_in[5];
  const float* Wproj  = (const float*)d_in[6];
  const float* b_proj = (const float*)d_in[7];
  const float* ln1_g  = (const float*)d_in[8];
  const float* ln1_b  = (const float*)d_in[9];
  const float* ln2_g  = (const float*)d_in[10];
  const float* ln2_b  = (const float*)d_in[11];
  const float* W1     = (const float*)d_in[12];
  const float* b1     = (const float*)d_in[13];
  const float* W2     = (const float*)d_in[14];
  const float* b2     = (const float*)d_in[15];

  const int BN = 2 * 2048, C = 1024, MLP = 4096;
  char* ws = (char*)d_ws;
  size_t off = 0;
  auto alloc = [&](size_t bytes) { char* p = ws + off; off += (bytes + 255) & ~(size_t)255; return p; };
  float*     x1     = (float*)    alloc((size_t)BN * C * 4);
  float*     x2     = (float*)    alloc((size_t)BN * C * 4);
  _Float16*  h1     = (_Float16*) alloc((size_t)BN * C * 2);
  _Float16*  h2     = (_Float16*) alloc((size_t)BN * C * 2);
  float*     afterA = (float*)    alloc((size_t)BN * 8 * 4);
  _Float16*  o16    = (_Float16*) alloc((size_t)BN * C * 2);
  _Float16*  qkv16  = (_Float16*) alloc((size_t)BN * 3 * C * 2);
  _Float16*  g16    = (_Float16*) alloc((size_t)BN * MLP * 2);
  _Float16*  Wqkv16 = (_Float16*) alloc((size_t)3 * C * C * 2);
  _Float16*  Wproj16= (_Float16*) alloc((size_t)C * C * 2);
  _Float16*  W116   = (_Float16*) alloc((size_t)MLP * C * 2);
  _Float16*  W216   = (_Float16*) alloc((size_t)C * MLP * 2);

  auto cvt = [&](const float* s, _Float16* d, int n) {
    k_cvt_f16<<<(n + 255) / 256, 256, 0, stream>>>(s, d, n);
  };
  cvt(Wqkv,  Wqkv16,  3 * C * C);
  cvt(Wproj, Wproj16, C * C);
  cvt(W1,    W116,    MLP * C);
  cvt(W2,    W216,    C * MLP);

  // 1) x1 = x + pos; h1 = LN1(x1); afterA = h1 @ lora_A^T
  k_add_ln<<<BN, 256, 0, stream>>>(x, pos, ln1_g, ln1_b, x1, h1, loraA, afterA, 1);

  // 2) qkv = h1 @ Wqkv^T + b + LoRA  -> f16
  k_gemm_wmma<<<dim3(BN / 32, (3 * C) / 512), 256, 0, stream>>>(
      h1, Wqkv16, b_qkv, BN, 3 * C, C, 1, afterA, loraB, nullptr, nullptr, qkv16);

  // 3) flash attention -> o16
  k_flash_attn<<<dim3(2048 / 64, 16, 2), 128, 0, stream>>>(qkv16, o16);

  // 4) x2 = x1 + o16 @ Wproj^T + b_proj
  k_gemm_wmma<<<dim3(BN / 32, C / 512), 256, 0, stream>>>(
      o16, Wproj16, b_proj, BN, C, C, 2, nullptr, nullptr, x1, x2, nullptr);

  // 5) h2 = LN2(x2)
  k_add_ln<<<BN, 256, 0, stream>>>(x2, nullptr, ln2_g, ln2_b, nullptr, h2, nullptr, nullptr, 2);

  // 6) g = gelu(h2 @ W1^T + b1)
  k_gemm_wmma<<<dim3(BN / 32, MLP / 512), 256, 0, stream>>>(
      h2, W116, b1, BN, MLP, C, 3, nullptr, nullptr, nullptr, nullptr, g16);

  // 7) out = x2 + g @ W2^T + b2
  k_gemm_wmma<<<dim3(BN / 32, C / 512), 256, 0, stream>>>(
      g16, W216, b2, BN, C, MLP, 4, nullptr, nullptr, x2, (float*)d_out, nullptr);
}
